// DeepSeekMoEBlock_49718541418558
// MI455X (gfx1250) — matrix operand order repaired and checked by reading, not compiled, not run on hardware
//
#include <hip/hip_runtime.h>
#include <hip/hip_bf16.h>

typedef __attribute__((ext_vector_type(16))) __bf16 v16bf;
typedef __attribute__((ext_vector_type(8)))  float  v8f;

union FragBF { v16bf v; unsigned int u[8]; };
union BPack  { __bf16 h[8]; uint4 u; };

__device__ __forceinline__ v8f wmma_bf16(const FragBF& a, const FragBF& b, v8f c) {
    return __builtin_amdgcn_wmma_f32_16x16x32_bf16(false, a.v, false, b.v, (short)0, c, false, false);
}

// Async copy of 32 contiguous bytes global(bf16) -> LDS, tracked by ASYNCcnt.
// The immediate offset is added to both the LDS and the global address.
__device__ __forceinline__ void async_copy32(unsigned lds_off, const void* gaddr) {
    asm volatile("global_load_async_to_lds_b128 %0, %1, off\n\t"
                 "global_load_async_to_lds_b128 %0, %1, off offset:16"
                 :: "v"(lds_off), "v"(gaddr) : "memory");
}
__device__ __forceinline__ void wait_async0() {
    asm volatile("s_wait_asynccnt 0x0" ::: "memory");
}

__device__ __forceinline__ float fast_silu(float x) {
    // x * sigmoid(x); v_rcp_f32 instead of the precise-division sequence
    return x * __builtin_amdgcn_rcpf(1.0f + __expf(-x));
}

// ---------------------------------------------------------------------------
// Kernel 0: convert x (f32) -> Xb (bf16)
// ---------------------------------------------------------------------------
__global__ void cvt_kernel(const float* __restrict__ X, __bf16* __restrict__ Xb, int n) {
    int i = (blockIdx.x * blockDim.x + threadIdx.x) * 4;
    if (i >= n) return;
    float4 v = *(const float4*)(X + i);
    union { __bf16 h[4]; uint2 u; } o;
    o.h[0] = (__bf16)v.x; o.h[1] = (__bf16)v.y; o.h[2] = (__bf16)v.z; o.h[3] = (__bf16)v.w;
    *(uint2*)(Xb + i) = o.u;
}

// ---------------------------------------------------------------------------
// Kernel 1: routing. logits = x @ Wg^T, fp32 softmax, top-4, cw[e*T + t]
// ---------------------------------------------------------------------------
__global__ void routing_kernel(const float* __restrict__ X, const float* __restrict__ Wg,
                               float* __restrict__ cw, int T, int D) {
    __shared__ float Xs[256][33];
    const int tid = threadIdx.x;
    const int t0  = blockIdx.x * 256;

    float acc[16];
#pragma unroll
    for (int e = 0; e < 16; ++e) acc[e] = 0.0f;

    const int c     = tid & 31;
    const int rbase = tid >> 5;

    for (int d0 = 0; d0 < D; d0 += 32) {
        __syncthreads();
#pragma unroll
        for (int rb = 0; rb < 32; ++rb) {
            int r = rb * 8 + rbase;
            Xs[r][c] = X[(size_t)(t0 + r) * D + d0 + c];
        }
        __syncthreads();
        for (int j = 0; j < 32; ++j) {
            float xv = Xs[tid][j];
            const float* wg = Wg + d0 + j;
#pragma unroll
            for (int e = 0; e < 16; ++e) acc[e] += xv * wg[(size_t)e * D];
        }
    }

    float mx = acc[0];
#pragma unroll
    for (int e = 1; e < 16; ++e) mx = fmaxf(mx, acc[e]);
    float p[16]; float se = 0.0f;
#pragma unroll
    for (int e = 0; e < 16; ++e) { p[e] = __expf(acc[e] - mx); se += p[e]; }
    float inv = 1.0f / se;   // once per token: keep precise
#pragma unroll
    for (int e = 0; e < 16; ++e) p[e] *= inv;

    // top-4, ties -> lowest index (matches lax.top_k)
    unsigned sel = 0u;
    for (int k = 0; k < 4; ++k) {
        int best = 0; float bv = -1.0f;
#pragma unroll
        for (int e = 0; e < 16; ++e) {
            if (!((sel >> e) & 1u) && p[e] > bv) { bv = p[e]; best = e; }
        }
        sel |= 1u << best;
    }

    const int t = t0 + tid;
#pragma unroll
    for (int e = 0; e < 16; ++e)
        cw[(size_t)e * T + t] = ((sel >> e) & 1u) ? p[e] : 0.0f;
}

// ---------------------------------------------------------------------------
// Kernel 2: H[t,f] = silu(X*W1^T) * (X*W3^T) * cw[t]   (bf16 WMMA, f32 acc)
//   Double-buffered K pipeline; A tile staged with global_load_async_to_lds.
// ---------------------------------------------------------------------------
__global__ void gemm1_kernel(const __bf16* __restrict__ Xb,
                             const float*  __restrict__ W1,
                             const float*  __restrict__ W3,
                             const float*  __restrict__ cwp,   // [T] or nullptr (==1.0)
                             __bf16* __restrict__ H,           // [T,F] bf16
                             int T, int D, int F) {
    __shared__ __bf16 As[2][128][40];
    __shared__ __bf16 Bs[2][2][64][40];

    const int tid      = threadIdx.x;
    const int rowBlock = blockIdx.y * 128;
    const int colBlock = blockIdx.x * 64;

    const int ar = tid >> 1;            // 0..127
    const int ac = (tid & 1) * 16;      // 0 / 16
    const int bn = tid >> 2;            // 0..63
    const int bc = (tid & 3) * 8;       // 0,8,16,24

    const int lane = tid & 31;
    const int wid  = tid >> 5;
    const int wm   = (wid >> 1) * 32;   // 0,32,64,96
    const int wn   = (wid & 1) * 32;    // 0,32

    v8f acc1[2][2], acc3[2][2];
#pragma unroll
    for (int mt = 0; mt < 2; ++mt)
#pragma unroll
        for (int nt = 0; nt < 2; ++nt) {
            acc1[mt][nt] = (v8f){0.f,0.f,0.f,0.f,0.f,0.f,0.f,0.f};
            acc3[mt][nt] = (v8f){0.f,0.f,0.f,0.f,0.f,0.f,0.f,0.f};
        }

    const int hi8  = (lane >> 4) * 8;
    const int arow = lane & 15;
    const int bcol = lane & 15;
    const int hk   = (lane >> 4) * 16;

    auto stage = [&](int buf, int k0) {
        // A: async global(bf16) -> LDS copy (ASYNCcnt)
        async_copy32((unsigned)(size_t)&As[buf][ar][ac],
                     Xb + (size_t)(rowBlock + ar) * D + k0 + ac);
        // B1/B3: f32 -> bf16 convert through VGPRs (LOADcnt + DScnt)
        const float* w1p = W1 + (size_t)(colBlock + bn) * D + k0 + bc;
        const float* w3p = W3 + (size_t)(colBlock + bn) * D + k0 + bc;
        float4 f0 = *(const float4*)(w1p);
        float4 f1 = *(const float4*)(w1p + 4);
        BPack pk;
        pk.h[0]=(__bf16)f0.x; pk.h[1]=(__bf16)f0.y; pk.h[2]=(__bf16)f0.z; pk.h[3]=(__bf16)f0.w;
        pk.h[4]=(__bf16)f1.x; pk.h[5]=(__bf16)f1.y; pk.h[6]=(__bf16)f1.z; pk.h[7]=(__bf16)f1.w;
        *(uint4*)&Bs[buf][0][bn][bc] = pk.u;
        f0 = *(const float4*)(w3p);
        f1 = *(const float4*)(w3p + 4);
        pk.h[0]=(__bf16)f0.x; pk.h[1]=(__bf16)f0.y; pk.h[2]=(__bf16)f0.z; pk.h[3]=(__bf16)f0.w;
        pk.h[4]=(__bf16)f1.x; pk.h[5]=(__bf16)f1.y; pk.h[6]=(__bf16)f1.z; pk.h[7]=(__bf16)f1.w;
        *(uint4*)&Bs[buf][1][bn][bc] = pk.u;
    };

    auto compute = [&](int buf) {
        FragBF afrag[2], b1frag[2], b3frag[2];
#pragma unroll
        for (int mt = 0; mt < 2; ++mt) {
            int r = wm + mt * 16 + arow;
#pragma unroll
            for (int i = 0; i < 8; ++i) {
                int K = ((i >> 2) << 4) + hi8 + ((i & 3) << 1);
                afrag[mt].u[i] = *(const unsigned int*)&As[buf][r][K];
            }
        }
#pragma unroll
        for (int nt = 0; nt < 2; ++nt) {
            int n = wn + nt * 16 + bcol;
#pragma unroll
            for (int i = 0; i < 8; ++i) {
                int K = ((i >> 2) << 4) + hi8 + ((i & 3) << 1);
                b1frag[nt].u[i] = *(const unsigned int*)&Bs[buf][0][n][K];
                b3frag[nt].u[i] = *(const unsigned int*)&Bs[buf][1][n][K];
            }
        }
#pragma unroll
        for (int mt = 0; mt < 2; ++mt)
#pragma unroll
            for (int nt = 0; nt < 2; ++nt) {
                acc1[mt][nt] = wmma_bf16(afrag[mt], b1frag[nt], acc1[mt][nt]);
                acc3[mt][nt] = wmma_bf16(afrag[mt], b3frag[nt], acc3[mt][nt]);
            }
    };

    stage(0, 0);
    wait_async0();
    __syncthreads();

    int cur = 0;
    for (int k0 = 0; k0 < D; k0 += 32) {
        const int nxt = cur ^ 1;
        const bool more = (k0 + 32) < D;
        if (more) stage(nxt, k0 + 32);
        compute(cur);
        if (more) wait_async0();
        __syncthreads();
        cur = nxt;
    }

    // epilogue: silu(a1)*a3*cw -> bf16 H  (uniform branch on cwp)
    const int hi  = lane >> 4;
    const int col = lane & 15;
    if (cwp) {
#pragma unroll
        for (int mt = 0; mt < 2; ++mt) {
            float wrow[8];
#pragma unroll
            for (int i = 0; i < 8; ++i)
                wrow[i] = cwp[rowBlock + wm + mt * 16 + hi * 8 + i];
#pragma unroll
            for (int nt = 0; nt < 2; ++nt)
#pragma unroll
                for (int i = 0; i < 8; ++i) {
                    int t = rowBlock + wm + mt * 16 + hi * 8 + i;
                    int f = colBlock + wn + nt * 16 + col;
                    float h = fast_silu(acc1[mt][nt][i]) * acc3[mt][nt][i] * wrow[i];
                    H[(size_t)t * F + f] = (__bf16)h;
                }
        }
    } else {
#pragma unroll
        for (int mt = 0; mt < 2; ++mt)
#pragma unroll
            for (int nt = 0; nt < 2; ++nt)
#pragma unroll
                for (int i = 0; i < 8; ++i) {
                    int t = rowBlock + wm + mt * 16 + hi * 8 + i;
                    int f = colBlock + wn + nt * 16 + col;
                    float h = fast_silu(acc1[mt][nt][i]) * acc3[mt][nt][i];
                    H[(size_t)t * F + f] = (__bf16)h;
                }
    }
}

// ---------------------------------------------------------------------------
// Kernel 3: OUT[t,n] (+)= H[t,:] @ W2[n,:]     (K = Fk)
// ---------------------------------------------------------------------------
__global__ void gemm2_kernel(const __bf16* __restrict__ H,
                             const float*  __restrict__ W2,
                             float* __restrict__ OUT,
                             int T, int D, int Fk, int accumulate) {
    __shared__ __bf16 As[2][128][40];
    __shared__ __bf16 Bs[2][64][40];

    const int tid      = threadIdx.x;
    const int rowBlock = blockIdx.y * 128;
    const int colBlock = blockIdx.x * 64;

    const int ar = tid >> 1;
    const int ac = (tid & 1) * 16;
    const int bn = tid >> 2;
    const int bc = (tid & 3) * 8;

    const int lane = tid & 31;
    const int wid  = tid >> 5;
    const int wm   = (wid >> 1) * 32;
    const int wn   = (wid & 1) * 32;

    v8f acc[2][2];
#pragma unroll
    for (int mt = 0; mt < 2; ++mt)
#pragma unroll
        for (int nt = 0; nt < 2; ++nt)
            acc[mt][nt] = (v8f){0.f,0.f,0.f,0.f,0.f,0.f,0.f,0.f};

    const int hi8  = (lane >> 4) * 8;
    const int arow = lane & 15;
    const int bcol = lane & 15;
    const int hk   = (lane >> 4) * 16;

    auto stage = [&](int buf, int k0) {
        async_copy32((unsigned)(size_t)&As[buf][ar][ac],
                     H + (size_t)(rowBlock + ar) * Fk + k0 + ac);
        const float* wp = W2 + (size_t)(colBlock + bn) * Fk + k0 + bc;
        float4 f0 = *(const float4*)(wp);
        float4 f1 = *(const float4*)(wp + 4);
        BPack pk;
        pk.h[0]=(__bf16)f0.x; pk.h[1]=(__bf16)f0.y; pk.h[2]=(__bf16)f0.z; pk.h[3]=(__bf16)f0.w;
        pk.h[4]=(__bf16)f1.x; pk.h[5]=(__bf16)f1.y; pk.h[6]=(__bf16)f1.z; pk.h[7]=(__bf16)f1.w;
        *(uint4*)&Bs[buf][bn][bc] = pk.u;
    };

    auto compute = [&](int buf) {
        FragBF afrag[2], bfrag[2];
#pragma unroll
        for (int mt = 0; mt < 2; ++mt) {
            int r = wm + mt * 16 + arow;
#pragma unroll
            for (int i = 0; i < 8; ++i) {
                int K = ((i >> 2) << 4) + hi8 + ((i & 3) << 1);
                afrag[mt].u[i] = *(const unsigned int*)&As[buf][r][K];
            }
        }
#pragma unroll
        for (int nt = 0; nt < 2; ++nt) {
            int n = wn + nt * 16 + bcol;
#pragma unroll
            for (int i = 0; i < 8; ++i) {
                int K = ((i >> 2) << 4) + hi8 + ((i & 3) << 1);
                bfrag[nt].u[i] = *(const unsigned int*)&Bs[buf][n][K];
            }
        }
#pragma unroll
        for (int mt = 0; mt < 2; ++mt)
#pragma unroll
            for (int nt = 0; nt < 2; ++nt)
                acc[mt][nt] = wmma_bf16(afrag[mt], bfrag[nt], acc[mt][nt]);
    };

    stage(0, 0);
    wait_async0();
    __syncthreads();

    int cur = 0;
    for (int k0 = 0; k0 < Fk; k0 += 32) {
        const int nxt = cur ^ 1;
        const bool more = (k0 + 32) < Fk;
        if (more) stage(nxt, k0 + 32);
        compute(cur);
        if (more) wait_async0();
        __syncthreads();
        cur = nxt;
    }

    const int hi  = lane >> 4;
    const int col = lane & 15;
    if (accumulate) {
#pragma unroll
        for (int mt = 0; mt < 2; ++mt)
#pragma unroll
            for (int nt = 0; nt < 2; ++nt)
#pragma unroll
                for (int i = 0; i < 8; ++i) {
                    int t = rowBlock + wm + mt * 16 + hi * 8 + i;
                    int n = colBlock + wn + nt * 16 + col;
                    OUT[(size_t)t * D + n] += acc[mt][nt][i];
                }
    } else {
#pragma unroll
        for (int mt = 0; mt < 2; ++mt)
#pragma unroll
            for (int nt = 0; nt < 2; ++nt)
#pragma unroll
                for (int i = 0; i < 8; ++i) {
                    int t = rowBlock + wm + mt * 16 + hi * 8 + i;
                    int n = colBlock + wn + nt * 16 + col;
                    OUT[(size_t)t * D + n] = acc[mt][nt][i];
                }
    }
}

// ---------------------------------------------------------------------------
extern "C" void kernel_launch(void* const* d_in, const int* in_sizes, int n_in,
                              void* d_out, int out_size, void* d_ws, size_t ws_size,
                              hipStream_t stream) {
    const float* X   = (const float*)d_in[0];
    const float* Wg  = (const float*)d_in[1];
    const float* W1  = (const float*)d_in[2];
    const float* W2  = (const float*)d_in[3];
    const float* W3  = (const float*)d_in[4];
    const float* W1s = (const float*)d_in[5];
    const float* W2s = (const float*)d_in[6];
    const float* W3s = (const float*)d_in[7];
    float* OUT = (float*)d_out;

    const int T = 2048, D = 2048, F = 1408, E = 16, FS = 2816;

    char* ws = (char*)d_ws;
    float*  cw = (float*)ws;                                        // [E][T] f32
    __bf16* Xb = (__bf16*)(ws + (size_t)E * T * sizeof(float));     // [T][D] bf16
    __bf16* H  = (__bf16*)(ws + (size_t)E * T * sizeof(float)
                              + (size_t)T * D * sizeof(__bf16));    // [T][FS] bf16 (reused)

    cvt_kernel<<<(T * D) / (256 * 4), 256, 0, stream>>>(X, Xb, T * D);
    routing_kernel<<<T / 256, 256, 0, stream>>>(X, Wg, cw, T, D);

    // shared expert (combine == 1), writes OUT
    gemm1_kernel<<<dim3(FS / 64, T / 128), 256, 0, stream>>>(Xb, W1s, W3s, nullptr, H, T, D, FS);
    gemm2_kernel<<<dim3(D / 64, T / 128), 256, 0, stream>>>(H, W2s, OUT, T, D, FS, 0);

    // routed experts: dense over all experts, combine weight zeroes unselected
    for (int e = 0; e < E; ++e) {
        gemm1_kernel<<<dim3(F / 64, T / 128), 256, 0, stream>>>(
            Xb, W1 + (size_t)e * F * D, W3 + (size_t)e * F * D, cw + (size_t)e * T, H, T, D, F);
        gemm2_kernel<<<dim3(D / 64, T / 128), 256, 0, stream>>>(
            H, W2 + (size_t)e * D * F, OUT, T, D, F, 1);
    }
}
